// GRUPredictor_7559142441116
// MI455X (gfx1250) — matrix operand order repaired and checked by reading, not compiled, not run on hardware
//
#include <hip/hip_runtime.h>
#include <hip/hip_bf16.h>

// ---------------- problem constants ----------------
#define T_SEQ 512
#define B_SZ  64
#define H_SZ  512
#define G3H   1536          // 3*H
#define NBLK  16            // persistent blocks (1 per WGP)
#define HS    32            // hidden units per block  (H / NBLK)
#define WROWS 96            // 3*HS gate rows per block
#define WSTR  520           // padded LDS row stride (bf16 elems), 16B-aligned rows
#define GS    132           // gates LDS row stride (floats)
#define NCLS  56

typedef __attribute__((ext_vector_type(16))) __bf16 v16bf;
typedef __attribute__((ext_vector_type(8)))  float  v8f;

union Frag  { v16bf v; uint4 q[2]; };
union H8u   { __bf16 h[8]; uint4 q; };
union F8u   { float  f[8]; float4 q4[2]; };

// LDS layout (bytes):
//   Whh (96x520 bf16)            @ 0
//   Wih (96x520 bf16)            @ 99,840
//   gates (64x132 f32)           @ 199,680
//   bih/bhh/w0 (3x96 f32)        @ 233,472
//   y_stage (64x520 bf16)        @ 234,624   <- filled by global_load_async_to_lds
constexpr int YS_LDS_OFF = WROWS*WSTR*2*2 + 64*GS*4 + 3*WROWS*4;   // 234,624
constexpr int SMEM_BYTES = YS_LDS_OFF + B_SZ*WSTR*2;               // 301,184 (<320KB WGP LDS)

// ---- fragment loaders (CDNA5 16-bit WMMA layouts, wave32) ----
// A 16x32 bf16: lane L -> row M=L%16, half=L/16; k = kk*32 + {half*8+0..7} U {16+half*8+0..7}
__device__ __forceinline__ v16bf load_a_glb(const __bf16* base, int row, int kk, int half) {
  const __bf16* p = base + (size_t)row * H_SZ + kk * 32 + half * 8;
  Frag f;
  f.q[0] = *(const uint4*)(p);
  f.q[1] = *(const uint4*)(p + 16);
  return f.v;
}
__device__ __forceinline__ v16bf load_a_lds(const __bf16* base, int row, int kk, int half) {
  const __bf16* p = base + row * WSTR + kk * 32 + half * 8;   // padded rows -> bank-skewed
  Frag f;
  f.q[0] = *(const uint4*)(p);
  f.q[1] = *(const uint4*)(p + 16);
  return f.v;
}
// B 32x16 bf16: lane L -> col N=L%16, half=L/16; k = kk*32 + {half*8+0..7} U {16+half*8+0..7}
__device__ __forceinline__ v16bf load_b_frag(const __bf16* ws, int wrow, int kk, int half) {
  const __bf16* p = ws + (size_t)wrow * WSTR + kk * 32 + half * 8;
  Frag f;
  f.q[0] = *(const uint4*)(p);
  f.q[1] = *(const uint4*)(p + 16);
  return f.v;
}

#define WMMA_BF16(A, Bf, C) \
  __builtin_amdgcn_wmma_f32_16x16x32_bf16(false, (A), false, (Bf), (short)0, (C), false, false)

// async-DMA one [64 x 512] bf16 timestep row-block into the padded LDS y_stage
__device__ __forceinline__ void issue_async_y(const __bf16* gsrc, int tid) {
#pragma unroll
  for (int i = 0; i < 16; ++i) {
    int idx = tid + (i << 8);                 // 0..4095 chunks of 16B
    int r = idx >> 6, c = idx & 63;
    unsigned voff = (unsigned)(r * (H_SZ * 2) + c * 16);          // global byte offset
    unsigned lds  = (unsigned)(YS_LDS_OFF + r * (WSTR * 2) + c * 16); // LDS byte offset
    asm volatile("global_load_async_to_lds_b128 %0, %1, %2"
                 :: "v"(lds), "v"(voff), "s"(gsrc) : "memory");
  }
}
__device__ __forceinline__ void wait_async() {
  asm volatile("s_wait_asynccnt 0x0" ::: "memory");
}

__global__ __launch_bounds__(256, 1)
void gru_layer_kernel(const unsigned short* __restrict__ y_prev_u,  // [T][B][H] bf16 (null for layer0)
                      const float* __restrict__ x,                  // [B][T] f32 (layer0 only)
                      const float* __restrict__ Wih,                // [1536][512] f32 (layers>=1)
                      const float* __restrict__ Whh,                // [1536][512] f32
                      const float* __restrict__ bih,                // [1536]
                      const float* __restrict__ bhh,                // [1536]
                      const float* __restrict__ Wih0,               // [1536] (layer0 only)
                      unsigned short* __restrict__ ys_u,            // [T][B][H] bf16 out (may be null)
                      unsigned short* __restrict__ hbf_u,           // [2][B][H] bf16 h double buffer
                      float* __restrict__ finals,                   // [B][H] f32 (offset by layer)
                      unsigned* __restrict__ ctr,                   // grid barrier counter (zeroed)
                      int layer0)
{
  extern __shared__ char smem[];
  __bf16* Whh_s = (__bf16*)smem;
  __bf16* Wih_s = Whh_s + WROWS * WSTR;
  float*  gates = (float*)(Wih_s + WROWS * WSTR);
  float*  bih_s = gates + 64 * GS;
  float*  bhh_s = bih_s + WROWS;
  float*  w0_s  = bhh_s + WROWS;
  __bf16* y_s   = (__bf16*)(smem + YS_LDS_OFF);

  const __bf16* y_prev = (const __bf16*)y_prev_u;
  __bf16* ys  = (__bf16*)ys_u;
  __bf16* hbf = (__bf16*)hbf_u;

  const int tid  = threadIdx.x;
  const int blk  = blockIdx.x;
  const bool hasI = (layer0 == 0);

  // kick off DMA of y[t=0] while we stage weights
  if (hasI) issue_async_y(y_prev, tid);

  // ---- stage weight slices into LDS (f32 -> bf16), gate rows {r,z,n} for our 32 units ----
  for (int i = tid; i < WROWS * H_SZ; i += 256) {
    int rr = i >> 9, k = i & (H_SZ - 1);
    int grow = (rr >> 5) * H_SZ + blk * HS + (rr & 31);
    Whh_s[rr * WSTR + k] = (__bf16)Whh[(size_t)grow * H_SZ + k];
    if (hasI) Wih_s[rr * WSTR + k] = (__bf16)Wih[(size_t)grow * H_SZ + k];
  }
  if (tid < WROWS) {
    int grow = (tid >> 5) * H_SZ + blk * HS + (tid & 31);
    bih_s[tid] = bih[grow];
    bhh_s[tid] = bhh[grow];
    w0_s[tid]  = layer0 ? Wih0[grow] : 0.f;
  }
  if (hasI) wait_async();
  __syncthreads();

  // wave tiling: 8 waves; m = batch tile (0..3), ng selects the gate-tile group.
  const int wave = tid >> 5, lane = tid & 31;
  const int m  = wave >> 1;
  const int ng = __builtin_amdgcn_readfirstlane(wave & 1);  // wave-uniform -> scalar branch
  const int nl = lane & 15, half = lane >> 4;
  const int arow = m * 16 + nl;

  // elementwise mapping: thread -> (batch eb, 8 hidden units from ejb)
  const int eb  = tid >> 2;
  const int ejb = (tid & 3) * 8;

  float h_old[8];
#pragma unroll
  for (int i = 0; i < 8; ++i) h_old[i] = 0.f;

  for (int t = 0; t < T_SEQ; ++t) {
    const __bf16* hcur = hbf + (size_t)(t & 1) * (B_SZ * H_SZ);
    const bool haveH = (t > 0);   // h(-1) == 0 -> skip hh GEMM at t=0

    v8f acc[5];
#pragma unroll
    for (int a = 0; a < 5; ++a) {
      v8f z8 = {0.f, 0.f, 0.f, 0.f, 0.f, 0.f, 0.f, 0.f};
      acc[a] = z8;
    }

    // ---- GEMM phase: gates = [y_t @ Wih^T] + [h @ Whh^T] via bf16 WMMA ----
    // B fragments grouped per pass so ds_loads overlap the WMMA chain.
#pragma unroll 2
    for (int kk = 0; kk < 16; ++kk) {
      if (ng == 0) {            // r/z combined tiles, cols 0..47
        if (haveH) {
          v16bf aH = load_a_glb(hcur, arow, kk, half);
          v16bf b0 = load_b_frag(Whh_s, 0 + nl,  kk, half);
          v16bf b1 = load_b_frag(Whh_s, 16 + nl, kk, half);
          v16bf b2 = load_b_frag(Whh_s, 32 + nl, kk, half);
          acc[0] = WMMA_BF16(aH, b0, acc[0]);
          acc[1] = WMMA_BF16(aH, b1, acc[1]);
          acc[2] = WMMA_BF16(aH, b2, acc[2]);
        }
        if (hasI) {
          v16bf aI = load_a_lds(y_s, arow, kk, half);
          v16bf b3 = load_b_frag(Wih_s, 0 + nl,  kk, half);
          v16bf b4 = load_b_frag(Wih_s, 16 + nl, kk, half);
          v16bf b5 = load_b_frag(Wih_s, 32 + nl, kk, half);
          acc[0] = WMMA_BF16(aI, b3, acc[0]);
          acc[1] = WMMA_BF16(aI, b4, acc[1]);
          acc[2] = WMMA_BF16(aI, b5, acc[2]);
        }
      } else {                  // z tile 48..63 combined; hn (hh rows 64..95); xn (ih rows 64..95)
        if (haveH) {
          v16bf aH = load_a_glb(hcur, arow, kk, half);
          v16bf b0 = load_b_frag(Whh_s, 48 + nl, kk, half);
          v16bf b1 = load_b_frag(Whh_s, 64 + nl, kk, half);
          v16bf b2 = load_b_frag(Whh_s, 80 + nl, kk, half);
          acc[0] = WMMA_BF16(aH, b0, acc[0]);
          acc[1] = WMMA_BF16(aH, b1, acc[1]);
          acc[2] = WMMA_BF16(aH, b2, acc[2]);
        }
        if (hasI) {
          v16bf aI = load_a_lds(y_s, arow, kk, half);
          v16bf b3 = load_b_frag(Wih_s, 48 + nl, kk, half);
          v16bf b4 = load_b_frag(Wih_s, 64 + nl, kk, half);
          v16bf b5 = load_b_frag(Wih_s, 80 + nl, kk, half);
          acc[0] = WMMA_BF16(aI, b3, acc[0]);
          acc[3] = WMMA_BF16(aI, b4, acc[3]);
          acc[4] = WMMA_BF16(aI, b5, acc[4]);
        }
      }
    }

    // C 16x16 f32: lane -> col N=lane%16; VGPR r -> row M = r + 8*(lane/16)
    if (ng == 0) {
#pragma unroll
      for (int r = 0; r < 8; ++r) {
        const int grow = (m * 16 + r + 8 * half) * GS + nl;
        gates[grow + 0]  = acc[0][r];
        gates[grow + 16] = acc[1][r];
        gates[grow + 32] = acc[2][r];
      }
    } else {
#pragma unroll
      for (int r = 0; r < 8; ++r) {
        const int grow = (m * 16 + r + 8 * half) * GS + nl;
        gates[grow + 48] = acc[0][r];
        gates[grow + 64] = acc[1][r];
        gates[grow + 80] = acc[2][r];
        if (!layer0) {
          gates[grow + 96]  = acc[3][r];
          gates[grow + 112] = acc[4][r];
        }
      }
    }
    __syncthreads();   // GEMM reads of y_s complete -> safe to start next DMA

    // DMA next timestep's input while we do the elementwise update + barrier
    if (hasI && (t + 1) < T_SEQ)
      issue_async_y(y_prev + (size_t)(t + 1) * (B_SZ * H_SZ), tid);

    // ---- elementwise GRU update for our 64x32 slice ----
    {
      const float xval = layer0 ? x[eb * T_SEQ + t] : 0.f;
      const float* g = gates + eb * GS;
      H8u hb; F8u hf;
#pragma unroll
      for (int i = 0; i < 8; ++i) {
        int j = ejb + i;
        float rp = g[j]      + bih_s[j]      + bhh_s[j];
        float zp = g[32 + j] + bih_s[32 + j] + bhh_s[32 + j];
        float hn = g[64 + j] + bhh_s[64 + j];
        float xn;
        if (layer0) {
          rp += xval * w0_s[j];
          zp += xval * w0_s[32 + j];
          xn  = xval * w0_s[64 + j] + bih_s[64 + j];
        } else {
          xn  = g[96 + j] + bih_s[64 + j];
        }
        float r  = 1.f / (1.f + __expf(-rp));
        float z  = 1.f / (1.f + __expf(-zp));
        float nv = tanhf(xn + r * hn);
        float hv = (1.f - z) * nv + z * h_old[i];
        h_old[i] = hv;
        hb.h[i]  = (__bf16)hv;
        hf.f[i]  = hv;
      }
      const size_t coff = (size_t)eb * H_SZ + blk * HS + ejb;
      *(uint4*)(hbf + (size_t)((t + 1) & 1) * (B_SZ * H_SZ) + coff) = hb.q;
      if (ys) *(uint4*)(ys + (size_t)t * (B_SZ * H_SZ) + coff) = hb.q;
      if (t == T_SEQ - 1) {
        *(float4*)(finals + coff)     = hf.q4[0];
        *(float4*)(finals + coff + 4) = hf.q4[1];
      }
    }

    // ---- grid barrier: all 16 blocks see the new h before next step ----
    if (t != T_SEQ - 1) {
      __builtin_amdgcn_fence(__ATOMIC_RELEASE, "agent");   // drain this thread's h stores
      __syncthreads();
      if (tid == 0) {
        __hip_atomic_fetch_add(ctr, 1u, __ATOMIC_RELEASE, __HIP_MEMORY_SCOPE_AGENT);
        const unsigned target = (unsigned)(t + 1) * (unsigned)NBLK;
        while (__hip_atomic_load(ctr, __ATOMIC_RELAXED, __HIP_MEMORY_SCOPE_AGENT) < target)
          __builtin_amdgcn_s_sleep(2);
      }
      if (hasI) wait_async();   // each wave drains its own y_{t+1} DMA before rendezvous
      __syncthreads();
      __builtin_amdgcn_fence(__ATOMIC_ACQUIRE, "agent");   // invalidate stale h lines
    }
  }
}

// ---------------- FC head: out = relu(h_out @ fc1^T + b1) @ fc2^T + b2 ----------------
__global__ __launch_bounds__(64)
void fc_head_kernel(const float* __restrict__ finals,  // [192][512]
                    const float* __restrict__ fc1w, const float* __restrict__ fc1b,
                    const float* __restrict__ fc2w, const float* __restrict__ fc2b,
                    float* __restrict__ out)           // [192][56]
{
  __shared__ float t20[20];
  const int row = blockIdx.x;
  const int tid = threadIdx.x;
  const float4* hp = (const float4*)(finals + (size_t)row * H_SZ);
  if (tid < 20) {
    const float4* wp = (const float4*)(fc1w + (size_t)tid * H_SZ);
    float s = 0.f;
#pragma unroll 4
    for (int i = 0; i < H_SZ / 4; ++i) {
      float4 a = hp[i], w = wp[i];
      s += a.x * w.x + a.y * w.y + a.z * w.z + a.w * w.w;
    }
    s += fc1b[tid];
    t20[tid] = s > 0.f ? s : 0.f;
  }
  __syncthreads();
  if (tid < NCLS) {
    float s = fc2b[tid];
#pragma unroll
    for (int k = 0; k < 20; ++k) s += t20[k] * fc2w[tid * 20 + k];
    out[(size_t)row * NCLS + tid] = s;
  }
}

// ---------------- workspace layout (requires ws_size >= ~68 MB) ----------------
constexpr size_t YS_BYTES  = (size_t)T_SEQ * B_SZ * H_SZ * 2;   // bf16 seq buffer
constexpr size_t OFF_YSA   = 0;
constexpr size_t OFF_YSB   = OFF_YSA + YS_BYTES;
constexpr size_t OFF_HBF   = OFF_YSB + YS_BYTES;                // 2*B*H bf16
constexpr size_t OFF_CTR   = OFF_HBF + (size_t)2 * B_SZ * H_SZ * 2;
constexpr size_t OFF_FIN   = OFF_CTR + 256;                     // 3*B*H f32

extern "C" void kernel_launch(void* const* d_in, const int* in_sizes, int n_in,
                              void* d_out, int out_size, void* d_ws, size_t ws_size,
                              hipStream_t stream) {
  const float* x         = (const float*)d_in[0];
  const float* W_ih0     = (const float*)d_in[1];
  const float* W_ih_rest = (const float*)d_in[2];
  const float* W_hh      = (const float*)d_in[3];
  const float* b_ih      = (const float*)d_in[4];
  const float* b_hh      = (const float*)d_in[5];
  const float* fc1w      = (const float*)d_in[6];
  const float* fc1b      = (const float*)d_in[7];
  const float* fc2w      = (const float*)d_in[8];
  const float* fc2b      = (const float*)d_in[9];
  float* out = (float*)d_out;
  char*  ws  = (char*)d_ws;

  unsigned short* ysA = (unsigned short*)(ws + OFF_YSA);
  unsigned short* ysB = (unsigned short*)(ws + OFF_YSB);
  unsigned short* hbf = (unsigned short*)(ws + OFF_HBF);
  unsigned*       ctr = (unsigned*)(ws + OFF_CTR);
  float*          fin = (float*)(ws + OFF_FIN);

  (void)in_sizes; (void)n_in; (void)out_size; (void)ws_size;

  // allow >64KB dynamic LDS (idempotent, deterministic)
  hipFuncSetAttribute((const void*)gru_layer_kernel,
                      hipFuncAttributeMaxDynamicSharedMemorySize, SMEM_BYTES);

  // zero grid-barrier counters (h state handled by t==0 hh-skip; no other init needed)
  hipMemsetAsync(ws + OFF_CTR, 0, 256, stream);

  // layer 0: input is rank-1 (IN=1), folded into elementwise phase
  gru_layer_kernel<<<NBLK, 256, SMEM_BYTES, stream>>>(
      nullptr, x, nullptr, W_hh, b_ih, b_hh, W_ih0,
      ysA, hbf, fin, ctr + 0, 1);
  // layer 1
  gru_layer_kernel<<<NBLK, 256, SMEM_BYTES, stream>>>(
      ysA, nullptr, W_ih_rest, W_hh + (size_t)1 * G3H * H_SZ,
      b_ih + G3H, b_hh + G3H, nullptr,
      ysB, hbf, fin + (size_t)B_SZ * H_SZ, ctr + 1, 0);
  // layer 2 (no ys output needed)
  gru_layer_kernel<<<NBLK, 256, SMEM_BYTES, stream>>>(
      ysB, nullptr, W_ih_rest + (size_t)G3H * H_SZ, W_hh + (size_t)2 * G3H * H_SZ,
      b_ih + 2 * G3H, b_hh + 2 * G3H, nullptr,
      nullptr, hbf, fin + (size_t)2 * B_SZ * H_SZ, ctr + 2, 0);

  // FC head over [3*64, 512] final hidden states
  fc_head_kernel<<<3 * B_SZ, 64, 0, stream>>>(fin, fc1w, fc1b, fc2w, fc2b, out);
}
